// BoxSampleEncoder_257698038143
// MI455X (gfx1250) — matrix operand order repaired and checked by reading, not compiled, not run on hardware
//
#include <hip/hip_runtime.h>
#include <hip/hip_bf16.h>

// Problem constants (match reference)
#define BB   4
#define NNQ  64
#define CC   256
#define HH   64
#define WW   64
#define ROIK 7
#define SSS  3
#define NINT 21                 // ROI * S
#define KDIM (CC * ROIK * ROIK) // 12544
#define MMR  (BB * NNQ)         // 256 rois
#define KSPLIT 8
#define KCHUNK (KDIM / KSPLIT)  // 1568 = 49 * 32
#define TROWS 20                // max pixel-window rows (boxes span <=18)
#define TCOLS 33                // 32 cols + 1 pad (bank-conflict avoidance)

typedef _Float16 v16h __attribute__((ext_vector_type(16)));
typedef _Float16 v8h  __attribute__((ext_vector_type(8)));
typedef float    v8f  __attribute__((ext_vector_type(8)));

// ---------------------------------------------------------------------------
// Kernel 1: ROI-align (bilinear 21x21 grid, 3x3 avg pool) -> A[m][k] in f16
// One block per box (8 waves). Each wave owns 32 channels; per channel it
// stages the <=18x18 pixel window with coalesced row loads into an LDS tile,
// then lanes compute the 49 pooled outputs from LDS.
// ---------------------------------------------------------------------------
__global__ __launch_bounds__(256) void roi_align_pack_kernel(
    const float* __restrict__ img,    // [B, C, H, W]
    const float* __restrict__ boxes,  // [B*N, 4] = x1,y1,x2,y2
    _Float16* __restrict__ A)         // [MMR, KDIM] row-major f16
{
    __shared__ int   sx0[NINT], sx1[NINT], sy0[NINT], sy1[NINT];
    __shared__ float swx[NINT], swy[NINT];
    __shared__ float tile[8][TROWS][TCOLS];

    const int m    = blockIdx.x;
    const int b    = m >> 6;              // m / N
    const int tid  = threadIdx.x;
    const int wave = tid >> 5;
    const int lane = tid & 31;

    const float nsx = (float)WW / (float)(WW - 1);
    const float nsy = (float)HH / (float)(HH - 1);
    const float x1 = boxes[m * 4 + 0] * nsx;
    const float y1 = boxes[m * 4 + 1] * nsy;
    const float x2 = boxes[m * 4 + 2] * nsx;
    const float y2 = boxes[m * 4 + 3] * nsy;

    if (tid < NINT) {
        const float t   = (float)tid * (1.0f / (NINT - 1));
        const float hsx = (x2 - x1) * (1.0f / (2 * NINT));
        const float gx  = x1 + hsx + t * (x2 - x1 - 2.0f * hsx);
        const float px  = fminf(fmaxf(gx * (float)(WW - 1), 0.0f), (float)(WW - 1));
        const float x0f = floorf(px);
        int x0 = (int)x0f; x0 = x0 < 0 ? 0 : (x0 > WW - 1 ? WW - 1 : x0);
        sx0[tid] = x0;
        sx1[tid] = (x0 + 1 < WW - 1) ? (x0 + 1) : (WW - 1);
        swx[tid] = px - x0f;
    } else if (tid >= 32 && tid < 32 + NINT) {
        const int   i   = tid - 32;
        const float t   = (float)i * (1.0f / (NINT - 1));
        const float hsy = (y2 - y1) * (1.0f / (2 * NINT));
        const float gy  = y1 + hsy + t * (y2 - y1 - 2.0f * hsy);
        const float py  = fminf(fmaxf(gy * (float)(HH - 1), 0.0f), (float)(HH - 1));
        const float y0f = floorf(py);
        int y0 = (int)y0f; y0 = y0 < 0 ? 0 : (y0 > HH - 1 ? HH - 1 : y0);
        sy0[i] = y0;
        sy1[i] = (y0 + 1 < HH - 1) ? (y0 + 1) : (HH - 1);
        swy[i] = py - y0f;
    }
    __syncthreads();

    // Pixel window (sample coords are monotonic in the grid index).
    const int xmin = sx0[0];
    const int ymin = sy0[0];
    int ncols = sx1[NINT - 1] - xmin + 1; if (ncols > 32)    ncols = 32;
    int nrows = sy1[NINT - 1] - ymin + 1; if (nrows > TROWS) nrows = TROWS;

    const float* __restrict__ imgb = img + (size_t)b * CC * (HH * WW);

    for (int cc = 0; cc < CC / 8; ++cc) {
        const int c = wave * (CC / 8) + cc;
        const float* __restrict__ imgc = imgb + (size_t)c * (HH * WW);

        // Stage window: one coalesced 32-lane load per row.
        if (lane < ncols) {
            const float* __restrict__ src = imgc + ymin * WW + xmin + lane;
            for (int r = 0; r < nrows; ++r)
                tile[wave][r][lane] = src[r * WW];
        }
        __syncthreads();

        // 49 pooled outputs per channel, lanes 0..31 then 0..16.
        _Float16* __restrict__ arow =
            A + (size_t)m * KDIM + (size_t)c * (ROIK * ROIK);
        for (int p = lane; p < ROIK * ROIK; p += 32) {
            const int ph = p / ROIK;
            const int pw = p - ph * ROIK;
            float acc = 0.0f;
            #pragma unroll
            for (int dy = 0; dy < SSS; ++dy) {
                const int   i   = ph * SSS + dy;
                const int   ry0 = sy0[i] - ymin;
                const int   ry1 = sy1[i] - ymin;
                const float wy  = swy[i];
                #pragma unroll
                for (int dx = 0; dx < SSS; ++dx) {
                    const int   j   = pw * SSS + dx;
                    const int   rx0 = sx0[j] - xmin;
                    const int   rx1 = sx1[j] - xmin;
                    const float wx  = swx[j];
                    const float v00 = tile[wave][ry0][rx0];
                    const float v01 = tile[wave][ry0][rx1];
                    const float v10 = tile[wave][ry1][rx0];
                    const float v11 = tile[wave][ry1][rx1];
                    const float top = v00 + (v01 - v00) * wx;
                    const float bot = v10 + (v11 - v10) * wx;
                    acc += top + (bot - top) * wy;
                }
            }
            arow[p] = (_Float16)(acc * (1.0f / (SSS * SSS)));
        }
        __syncthreads();
    }
}

// ---------------------------------------------------------------------------
// Kernel 2: pack conv_w (f32 [O,C,7,7]) -> Bp f16 [O][KDIM] row-major
// ---------------------------------------------------------------------------
__global__ __launch_bounds__(256) void pack_b_kernel(
    const float* __restrict__ conv_w, _Float16* __restrict__ Bp, int total)
{
    for (int i = blockIdx.x * blockDim.x + threadIdx.x; i < total;
         i += gridDim.x * blockDim.x) {
        Bp[i] = (_Float16)conv_w[i];
    }
}

// ---------------------------------------------------------------------------
// Kernel 3: bias[m][o] = conv_b + cxcywh_b + posenc_b + base + pos_enc
// ---------------------------------------------------------------------------
__global__ __launch_bounds__(256) void bias_kernel(
    const float* __restrict__ boxes,
    const float* __restrict__ conv_b,
    const float* __restrict__ cxcywh_w, const float* __restrict__ cxcywh_b,
    const float* __restrict__ posenc_w, const float* __restrict__ posenc_b,
    float* __restrict__ bias)
{
    __shared__ float pe[CC + 2];
    const int m   = blockIdx.x;
    const int tid = threadIdx.x;

    const float x1 = boxes[m * 4 + 0], y1 = boxes[m * 4 + 1];
    const float x2 = boxes[m * 4 + 2], y2 = boxes[m * 4 + 3];
    const float cx = 0.5f * (x1 + x2), cy = 0.5f * (y1 + y2);
    const float w  = x2 - x1,          h  = y2 - y1;

    // pe_tok = [y_enc(128), x_enc(128), h, w]
    {
        const int   half  = CC / 2;                 // 128
        const float coord = (tid < half) ? cy : cx;
        const int   idx   = tid & (half - 1);
        const int   j     = idx >> 1;
        // p = coord * 2*pi / 10000^(j/64)
        const float p = coord * 6.283185307179586f *
                        __expf(-(float)j * (1.0f / 64.0f) * 9.210340371976184f);
        pe[tid] = (idx & 1) ? __cosf(p) : __sinf(p);
        if (tid == 0) { pe[CC + 0] = h; pe[CC + 1] = w; }
    }
    __syncthreads();

    const int o = tid;
    float s = conv_b[o] + cxcywh_b[o] + posenc_b[o];
    s += cx * cxcywh_w[o * 4 + 0] + cy * cxcywh_w[o * 4 + 1] +
         w  * cxcywh_w[o * 4 + 2] + h  * cxcywh_w[o * 4 + 3];
    const float* __restrict__ pw = posenc_w + (size_t)o * (CC + 2);
    float acc = 0.0f;
    #pragma unroll 4
    for (int j = 0; j < CC + 2; ++j) acc += pe[j] * pw[j];
    bias[(size_t)m * CC + o] = s + acc;
}

// ---------------------------------------------------------------------------
// Kernel 4: split-K WMMA GEMM.  partial[ks][M][N] = A[:, kslice] * B^T
// grid = (8 tile-blocks, KSPLIT); 8 waves/block; each wave = 16(M) x 64(N)
// strip, 49 k-steps of V_WMMA_F32_16X16X32_F16.  512 waves total.
// ---------------------------------------------------------------------------
__global__ __launch_bounds__(256) void wmma_gemm_kernel(
    const _Float16* __restrict__ A,   // [MMR][KDIM]
    const _Float16* __restrict__ Bp,  // [CC][KDIM]
    float* __restrict__ partial)      // [KSPLIT][MMR][CC]
{
    const int wave = threadIdx.x >> 5;
    const int lane = threadIdx.x & 31;
    const int gt   = blockIdx.x * 8 + wave;  // 0..63 wave tiles
    const int tm   = gt >> 2;                // M tile: 0..15 (x16 rows)
    const int tn4  = gt & 3;                 // N quad: 0..3  (x64 cols)
    const int g    = lane >> 4;              // lane half-group
    const int r    = lane & 15;              // row/col within tile
    const int kbeg = blockIdx.y * KCHUNK;

    // A fragment rows: lane holds matrix row (lane&15); per k-step the 16
    // halves are contiguous: [k0+g*16, +16), loaded as two 8-half runs.
    const _Float16* __restrict__ arow = A + (size_t)(tm * 16 + r) * KDIM;
    // B fragment: lane holds output column n=(lane&15); 16 contiguous halves
    // at [k0 + g*16, +16) of Bp row n.
    const _Float16* __restrict__ br0 = Bp + (size_t)(tn4 * 64 +  0 + r) * KDIM;
    const _Float16* __restrict__ br1 = Bp + (size_t)(tn4 * 64 + 16 + r) * KDIM;
    const _Float16* __restrict__ br2 = Bp + (size_t)(tn4 * 64 + 32 + r) * KDIM;
    const _Float16* __restrict__ br3 = Bp + (size_t)(tn4 * 64 + 48 + r) * KDIM;

    v8f acc0 = {}, acc1 = {}, acc2 = {}, acc3 = {};

    for (int k0 = kbeg; k0 < kbeg + KCHUNK; k0 += 32) {
        __builtin_prefetch(arow + k0 + 512, 0, 1);  // global_prefetch_b8

        const v8h alo = *(const v8h*)(arow + k0 + g * 16);
        const v8h ahi = *(const v8h*)(arow + k0 + g * 16 + 8);
        const v16h a = __builtin_shufflevector(alo, ahi,
            0, 1, 2, 3, 4, 5, 6, 7, 8, 9, 10, 11, 12, 13, 14, 15);

        const v16h b0 = *(const v16h*)(br0 + k0 + g * 16);
        const v16h b1 = *(const v16h*)(br1 + k0 + g * 16);
        const v16h b2 = *(const v16h*)(br2 + k0 + g * 16);
        const v16h b3 = *(const v16h*)(br3 + k0 + g * 16);

        acc0 = __builtin_amdgcn_wmma_f32_16x16x32_f16(false, a, false, b0,
                                                      (short)0, acc0, false, false);
        acc1 = __builtin_amdgcn_wmma_f32_16x16x32_f16(false, a, false, b1,
                                                      (short)0, acc1, false, false);
        acc2 = __builtin_amdgcn_wmma_f32_16x16x32_f16(false, a, false, b2,
                                                      (short)0, acc2, false, false);
        acc3 = __builtin_amdgcn_wmma_f32_16x16x32_f16(false, a, false, b3,
                                                      (short)0, acc3, false, false);
    }

    // D layout: VGPR v -> m = v + 8*g, n = r.
    float* __restrict__ pout = partial + (size_t)blockIdx.y * (MMR * CC);
    const int nb = tn4 * 64;
    #pragma unroll
    for (int v = 0; v < 8; ++v) {
        const int    mG   = tm * 16 + v + 8 * g;
        const size_t base = (size_t)mG * CC + nb + r;
        pout[base +  0] = acc0[v];
        pout[base + 16] = acc1[v];
        pout[base + 32] = acc2[v];
        pout[base + 48] = acc3[v];
    }
}

// ---------------------------------------------------------------------------
// Kernel 5: deterministic reduction  out = bias + sum_ks partial[ks]
// ---------------------------------------------------------------------------
__global__ __launch_bounds__(256) void reduce_kernel(
    const float* __restrict__ partial, const float* __restrict__ bias,
    float* __restrict__ out)
{
    const int idx = blockIdx.x * 256 + threadIdx.x;
    float s = bias[idx];
    #pragma unroll
    for (int ks = 0; ks < KSPLIT; ++ks)
        s += partial[(size_t)ks * (MMR * CC) + idx];
    out[idx] = s;
}

// ---------------------------------------------------------------------------
// Launch
// ---------------------------------------------------------------------------
extern "C" void kernel_launch(void* const* d_in, const int* in_sizes, int n_in,
                              void* d_out, int out_size, void* d_ws, size_t ws_size,
                              hipStream_t stream) {
    const float* img      = (const float*)d_in[0];
    const float* boxes    = (const float*)d_in[1];
    const float* conv_w   = (const float*)d_in[2];
    const float* conv_b   = (const float*)d_in[3];
    const float* cxcywh_w = (const float*)d_in[4];
    const float* cxcywh_b = (const float*)d_in[5];
    const float* posenc_w = (const float*)d_in[6];
    const float* posenc_b = (const float*)d_in[7];
    float* out = (float*)d_out;

    // Workspace: A f16 (6.4MB) | B f16 (6.4MB) | bias f32 (256KB) | partial (2MB)
    char* ws = (char*)d_ws;
    const size_t aBytes    = (size_t)MMR * KDIM * sizeof(_Float16);
    const size_t bBytes    = (size_t)CC  * KDIM * sizeof(_Float16);
    const size_t biasBytes = (size_t)MMR * CC * sizeof(float);
    _Float16* Af      = (_Float16*)(ws);
    _Float16* Bf      = (_Float16*)(ws + aBytes);
    float*    bias    = (float*)   (ws + aBytes + bBytes);
    float*    partial = (float*)   (ws + aBytes + bBytes + biasBytes);

    roi_align_pack_kernel<<<MMR, 256, 0, stream>>>(img, boxes, Af);
    pack_b_kernel<<<2048, 256, 0, stream>>>(conv_w, Bf, CC * KDIM);
    bias_kernel<<<MMR, 256, 0, stream>>>(boxes, conv_b, cxcywh_w, cxcywh_b,
                                         posenc_w, posenc_b, bias);
    wmma_gemm_kernel<<<dim3(8, KSPLIT), 256, 0, stream>>>(Af, Bf, partial);
    reduce_kernel<<<(MMR * CC) / 256, 256, 0, stream>>>(partial, bias, out);
}
